// Block_54382875902076
// MI455X (gfx1250) — matrix operand order repaired and checked by reading, not compiled, not run on hardware
//
#include <hip/hip_runtime.h>
#include <math.h>

typedef __attribute__((ext_vector_type(16))) __bf16 bf16x16;
typedef __attribute__((ext_vector_type(8)))  float  f32x8;

#define LN_EPS 1e-6f

// Load a 16-element bf16 fragment from two contiguous 16B runs
__device__ __forceinline__ bf16x16 load_frag_pair(const __bf16* p0, const __bf16* p1) {
    union { bf16x16 v; uint4 u[2]; } r;
    r.u[0] = *(const uint4*)p0;   // elems 0..7
    r.u[1] = *(const uint4*)p1;   // elems 8..15
    return r.v;
}

// ---------------------------------------------------------------------------
// f32 -> bf16 elementwise convert
// ---------------------------------------------------------------------------
__global__ void f32_to_bf16_kernel(const float* __restrict__ in,
                                   __bf16* __restrict__ out, int n) {
    int i = blockIdx.x * blockDim.x + threadIdx.x;
    if (i < n) out[i] = (__bf16)in[i];
}

// ---------------------------------------------------------------------------
// f32 [K][N] -> bf16 transposed [N][K]  (one-time weight prep; coalesced read)
// ---------------------------------------------------------------------------
__global__ void f32_to_bf16_T_kernel(const float* __restrict__ in,
                                     __bf16* __restrict__ out, int K, int N) {
    int i = blockIdx.x * blockDim.x + threadIdx.x;
    if (i < K * N) {
        int k = i / N, n = i - k * N;
        out[(size_t)n * K + k] = (__bf16)in[i];
    }
}

// ---------------------------------------------------------------------------
// LayerNorm (no bias): one 256-thread block per row of E=512, bf16 output
// ---------------------------------------------------------------------------
__global__ void ln_bf16_kernel(const float* __restrict__ x,
                               const float* __restrict__ g,
                               __bf16* __restrict__ out, int E) {
    const int row = blockIdx.x;
    const int tid = threadIdx.x;              // blockDim.x == 256, E == 512
    __shared__ float red[256];

    const float* xr = x + (size_t)row * E;
    float a0 = xr[tid];
    float a1 = xr[tid + 256];

    red[tid] = a0 + a1;
    __syncthreads();
    for (int off = 128; off > 0; off >>= 1) {
        if (tid < off) red[tid] += red[tid + off];
        __syncthreads();
    }
    float mu = red[0] * (1.0f / 512.0f);
    __syncthreads();

    float d0 = a0 - mu, d1 = a1 - mu;
    red[tid] = d0 * d0 + d1 * d1;
    __syncthreads();
    for (int off = 128; off > 0; off >>= 1) {
        if (tid < off) red[tid] += red[tid + off];
        __syncthreads();
    }
    float var = red[0] * (1.0f / 512.0f);
    float rs = rsqrtf(var + LN_EPS);

    out[(size_t)row * E + tid]       = (__bf16)(d0 * rs * g[tid]);
    out[(size_t)row * E + tid + 256] = (__bf16)(d1 * rs * g[tid + 256]);
}

// ---------------------------------------------------------------------------
// Branchless tanh-approx GELU, built on v_exp_f32
// ---------------------------------------------------------------------------
__device__ __forceinline__ float gelu_tanh(float x) {
    const float k0 = 0.7978845608028654f;     // sqrt(2/pi)
    const float k1 = 0.044715f;
    float u = k0 * (x + k1 * x * x * x);
    float e = __expf(-2.0f * fabsf(u));
    float th = (1.0f - e) / (1.0f + e);
    th = copysignf(th, u);
    return 0.5f * x * (1.0f + th);
}

// ---------------------------------------------------------------------------
// bf16 WMMA GEMM, LDS-free: C[M,N] = A[M,K] (row-major) x B (given TRANSPOSED,
// Bt[N][K] row-major). One wave per 64(M) x 16(N) strip, 4 WMMA per k-step.
// All fragment loads are contiguous 16B/32B vector loads:
//   A frag (lane mn,hlf): rows m0+t*16+mn, runs [kb8..kb8+7] and [16+kb8..]
//   B frag (lane mn,hlf): Bt row n0+mn, runs [k0+kb8..+7] and [k0+16+kb8..+7]
// Epilogue: optional GELU, f32 residual add, f32/bf16 store, and optional
// per-head-transposed bf16 store (for V: out[b][n][s], s = m % tS).
// ---------------------------------------------------------------------------
__global__ __launch_bounds__(32)
void gemm_bf16_wmma(const __bf16* __restrict__ A, const __bf16* __restrict__ Bt,
                    int M, int N, int K,
                    float* __restrict__ Cf, __bf16* __restrict__ Cb,
                    __bf16* __restrict__ CbT, int tS,
                    const float* __restrict__ resid, int do_gelu) {
    const int lane = threadIdx.x;
    const int mn   = lane & 15;       // N index (B/C layout), M index (A layout)
    const int hlf  = lane >> 4;       // half-wave selector
    const int n0   = blockIdx.x * 16;
    const int m0   = blockIdx.y * 64;
    const int kb8  = hlf * 8;

    f32x8 acc[4];
    #pragma unroll
    for (int t = 0; t < 4; ++t) acc[t] = (f32x8){0.f,0.f,0.f,0.f,0.f,0.f,0.f,0.f};

    const __bf16* btRow = Bt + (size_t)(n0 + mn) * K + kb8;
    const __bf16* aRow0 = A + (size_t)(m0 + mn) * K + kb8;   // tile t adds t*16*K

    for (int k0 = 0; k0 < K; k0 += 32) {
        // B fragment: two contiguous 16B loads from transposed weights
        bf16x16 b = load_frag_pair(btRow + k0, btRow + k0 + 16);

        if (k0 + 32 < K) {
            __builtin_prefetch(btRow + k0 + 32, 0, 3);       // global_prefetch_b8
            __builtin_prefetch(aRow0 + k0 + 32, 0, 3);
        }

        #pragma unroll
        for (int t = 0; t < 4; ++t) {
            const __bf16* ap = aRow0 + (size_t)t * 16 * K + k0;
            bf16x16 a = load_frag_pair(ap, ap + 16);
            acc[t] = __builtin_amdgcn_wmma_f32_16x16x32_bf16(false, a, false, b,
                                                             (short)0, acc[t], false, false);
        }
    }

    // Epilogue: C layout lane mn = N, VGPR r = M row (r + 8*hlf)
    #pragma unroll
    for (int t = 0; t < 4; ++t) {
        #pragma unroll
        for (int r = 0; r < 8; ++r) {
            const int m = m0 + t * 16 + r + 8 * hlf;
            const int n = n0 + mn;
            const size_t idx = (size_t)m * N + n;
            float v = acc[t][r];
            if (do_gelu) v = gelu_tanh(v);
            if (resid)   v += resid[idx];
            if (Cf)  Cf[idx] = v;
            if (Cb)  Cb[idx] = (__bf16)v;
            if (CbT) {
                const int bb = m / tS, s = m % tS;           // tS is a power of two
                CbT[((size_t)bb * N + n) * tS + s] = (__bf16)v;
            }
        }
    }
}

// ---------------------------------------------------------------------------
// Causal flash attention, one wave per (b, h, 16-row query tile).
// q/k: bf16 [B, S, H*D] (D=64); v: bf16 TRANSPOSED [B, H*D, S].
// All Q/K/V fragment loads are contiguous 16B/32B vector loads; the P tile is
// written to LDS directly in A-fragment order so the read side is one 32B
// vector load per lane.
// ---------------------------------------------------------------------------
__global__ __launch_bounds__(32)
void attn_flash_kernel(const __bf16* __restrict__ qb, const __bf16* __restrict__ kb,
                       const __bf16* __restrict__ vt, __bf16* __restrict__ ob,
                       int S, int E) {
    const int lane = threadIdx.x;
    const int mn   = lane & 15;
    const int hlf  = lane >> 4;
    const int qt = blockIdx.x, h = blockIdx.y, b = blockIdx.z;
    const int i0 = qt * 16;
    const size_t baseRow = (size_t)b * S;
    const int hoff = h * 64;
    const int kb8  = hlf * 8;
    const int ko   = hlf * 16;

    // Q fragments for K-dim 0..31 and 32..63 (vector loads, held in registers)
    const __bf16* qp = qb + (baseRow + i0 + mn) * (size_t)E + hoff;
    bf16x16 aq0 = load_frag_pair(qp + kb8,      qp + 16 + kb8);
    bf16x16 aq1 = load_frag_pair(qp + 32 + kb8, qp + 48 + kb8);

    float mi[8], li[8];
    f32x8 accO[4];
    #pragma unroll
    for (int r = 0; r < 8; ++r) { mi[r] = -__builtin_inff(); li[r] = 0.f; }
    #pragma unroll
    for (int t = 0; t < 4; ++t) accO[t] = (f32x8){0.f,0.f,0.f,0.f,0.f,0.f,0.f,0.f};

    // P tile in A-fragment order: [dest lane][16 elems]
    __shared__ __align__(32) __bf16 ldsP[32 * 16];

    const int qmaxRow = i0 + 15;
    const int nchunks = qmaxRow / 32 + 1;    // 32 kv columns per chunk

    for (int c = 0; c < nchunks; ++c) {
        float stile[2][8];
        #pragma unroll
        for (int t = 0; t < 2; ++t) {
            const int j0 = c * 32 + t * 16;
            f32x8 sA = (f32x8){0.f,0.f,0.f,0.f,0.f,0.f,0.f,0.f};
            if (j0 <= qmaxRow) {             // wave-uniform: EXEC stays full
                // K fragments: contraction = head-dim = fastest axis -> vector loads
                const __bf16* kp = kb + (baseRow + j0 + mn) * (size_t)E + hoff;
                bf16x16 bk0 = load_frag_pair(kp + kb8, kp + 16 + kb8);        // dims 0..31
                bf16x16 bk1 = load_frag_pair(kp + 32 + kb8, kp + 48 + kb8);   // dims 32..63
                sA = __builtin_amdgcn_wmma_f32_16x16x32_bf16(false, aq0, false, bk0,
                                                             (short)0, sA, false, false);
                sA = __builtin_amdgcn_wmma_f32_16x16x32_bf16(false, aq1, false, bk1,
                                                             (short)0, sA, false, false);
            }
            #pragma unroll
            for (int r = 0; r < 8; ++r) {
                float v = sA[r] * 0.125f;                // 1/sqrt(64)
                const int qrow = i0 + r + 8 * hlf;
                const int kcol = j0 + mn;
                if (j0 > qmaxRow || kcol > qrow) v = -__builtin_inff();
                stile[t][r] = v;
            }
        }

        // Online softmax update per row (rows live across 16-lane half-waves)
        const int g0 = mn >> 3;              // fragment-order store mapping
        #pragma unroll
        for (int r = 0; r < 8; ++r) {
            float mx = fmaxf(stile[0][r], stile[1][r]);
            mx = fmaxf(mx, __shfl_xor(mx, 1, 32));
            mx = fmaxf(mx, __shfl_xor(mx, 2, 32));
            mx = fmaxf(mx, __shfl_xor(mx, 4, 32));
            mx = fmaxf(mx, __shfl_xor(mx, 8, 32));
            float mnew  = fmaxf(mi[r], mx);
            float alpha = __expf(mi[r] - mnew);
            float p0 = __expf(stile[0][r] - mnew);
            float p1 = __expf(stile[1][r] - mnew);
            float rs = p0 + p1;
            rs += __shfl_xor(rs, 1, 32);
            rs += __shfl_xor(rs, 2, 32);
            rs += __shfl_xor(rs, 4, 32);
            rs += __shfl_xor(rs, 8, 32);
            li[r] = li[r] * alpha + rs;
            mi[r] = mnew;
            // scatter into A-fragment order: elem (m, kk): dest lane = m+16*((kk>>3)&1),
            // elem idx = ((kk>>3)>>1)*8 + (kk&7).  p0: kk=mn, p1: kk=16+mn (same dest lane)
            const int m  = r + 8 * hlf;
            const int dl = m + 16 * g0;
            ldsP[dl * 16 + (mn & 7)]     = (__bf16)p0;
            ldsP[dl * 16 + 8 + (mn & 7)] = (__bf16)p1;
            #pragma unroll
            for (int t = 0; t < 4; ++t) accO[t][r] *= alpha;
        }
        __syncthreads();

        // P fragment: one contiguous 32B LDS read per lane
        bf16x16 pf = *(const bf16x16*)(ldsP + lane * 16);

        // O += P x V: V transposed [B][E][S] -> contraction (kv pos) is fastest axis
        #pragma unroll
        for (int t = 0; t < 4; ++t) {
            const __bf16* vp = vt + ((size_t)b * E + hoff + t * 16 + mn) * S + c * 32 + kb8;
            bf16x16 vf = load_frag_pair(vp, vp + 16);
            accO[t] = __builtin_amdgcn_wmma_f32_16x16x32_bf16(false, pf, false, vf,
                                                              (short)0, accO[t], false, false);
        }
        __syncthreads();
    }

    // Writeout: O / l
    #pragma unroll
    for (int t = 0; t < 4; ++t) {
        #pragma unroll
        for (int r = 0; r < 8; ++r) {
            float v = accO[t][r] / li[r];
            ob[(baseRow + i0 + r + 8 * hlf) * (size_t)E + hoff + t * 16 + mn] = (__bf16)v;
        }
    }
}

// ---------------------------------------------------------------------------
// Host-side orchestration
// ---------------------------------------------------------------------------
extern "C" void kernel_launch(void* const* d_in, const int* in_sizes, int n_in,
                              void* d_out, int out_size, void* d_ws, size_t ws_size,
                              hipStream_t stream) {
    (void)in_sizes; (void)n_in; (void)out_size; (void)ws_size;

    constexpr int B = 4, S = 2048, E = 512, H = 8;
    constexpr int M = B * S;                 // 8192 rows
    constexpr int E4 = 4 * E;                // 2048

    const float* x    = (const float*)d_in[0];
    const float* wq   = (const float*)d_in[1];
    const float* wk   = (const float*)d_in[2];
    const float* wv   = (const float*)d_in[3];
    const float* wo1  = (const float*)d_in[4];
    const float* wo2  = (const float*)d_in[5];
    const float* w1   = (const float*)d_in[6];
    const float* w2   = (const float*)d_in[7];
    const float* ln1g = (const float*)d_in[8];
    const float* ln2g = (const float*)d_in[9];
    float* out = (float*)d_out;

    char* ws = (char*)d_ws;
    // workspace layout (bytes); weights stored TRANSPOSED [N][K] bf16
    const size_t OFF_WQ   = 0;                       // 512*512 bf16
    const size_t OFF_WK   = OFF_WQ  + 524288;
    const size_t OFF_WV   = OFF_WK  + 524288;
    const size_t OFF_WO1  = OFF_WV  + 524288;
    const size_t OFF_WO2  = OFF_WO1 + 524288;
    const size_t OFF_W1   = OFF_WO2 + 524288;        // [2048][512] bf16
    const size_t OFF_W2   = OFF_W1  + 2097152;       // [512][2048] bf16
    const size_t OFF_H1   = OFF_W2  + 2097152;       // 8192*512 bf16
    const size_t OFF_Q    = OFF_H1  + 8388608;
    const size_t OFF_K    = OFF_Q   + 8388608;
    const size_t OFF_VT   = OFF_K   + 8388608;       // V transposed [B][E][S] bf16
    const size_t OFF_ATT  = OFF_VT  + 8388608;       // attention out bf16
    const size_t OFF_P1   = OFF_ATT + 8388608;       // after wo1, bf16
    const size_t OFF_RES1 = OFF_P1  + 8388608;       // residual1 f32 (16MB)
    const size_t OFF_H2   = OFF_RES1 + 16777216;     // LN2 out bf16
    const size_t OFF_MLP1 = OFF_H2  + 8388608;       // 8192*2048 bf16 (32MB)

    __bf16* wqt  = (__bf16*)(ws + OFF_WQ);
    __bf16* wkt  = (__bf16*)(ws + OFF_WK);
    __bf16* wvt  = (__bf16*)(ws + OFF_WV);
    __bf16* wo1t = (__bf16*)(ws + OFF_WO1);
    __bf16* wo2t = (__bf16*)(ws + OFF_WO2);
    __bf16* w1t  = (__bf16*)(ws + OFF_W1);
    __bf16* w2t  = (__bf16*)(ws + OFF_W2);
    __bf16* h1   = (__bf16*)(ws + OFF_H1);
    __bf16* qbuf = (__bf16*)(ws + OFF_Q);
    __bf16* kbuf = (__bf16*)(ws + OFF_K);
    __bf16* vtb  = (__bf16*)(ws + OFF_VT);
    __bf16* attn = (__bf16*)(ws + OFF_ATT);
    __bf16* p1   = (__bf16*)(ws + OFF_P1);
    float*  res1 = (float*) (ws + OFF_RES1);
    __bf16* h2   = (__bf16*)(ws + OFF_H2);
    __bf16* mlp1 = (__bf16*)(ws + OFF_MLP1);

    // 1) convert weights to bf16, TRANSPOSED ([K][N] f32 -> [N][K] bf16)
    auto cvtT = [&](const float* src, __bf16* dst, int K, int N) {
        int n = K * N;
        f32_to_bf16_T_kernel<<<(n + 255) / 256, 256, 0, stream>>>(src, dst, K, N);
    };
    cvtT(wq,  wqt,  E, E);
    cvtT(wk,  wkt,  E, E);
    cvtT(wv,  wvt,  E, E);
    cvtT(wo1, wo1t, E, E);   // [H,D,E] row-major == [E,E] KxN view
    cvtT(wo2, wo2t, E, E);
    cvtT(w1,  w1t,  E, E4);
    cvtT(w2,  w2t,  E4, E);

    // 2) LN1
    ln_bf16_kernel<<<M, 256, 0, stream>>>(x, ln1g, h1, E);

    // 3) QKV projections  (grid: N/16 x M/64); V written transposed per batch
    dim3 gE(E / 16, M / 64);                 // (32, 128)
    gemm_bf16_wmma<<<gE, 32, 0, stream>>>(h1, wqt, M, E, E, nullptr, qbuf, nullptr, S, nullptr, 0);
    gemm_bf16_wmma<<<gE, 32, 0, stream>>>(h1, wkt, M, E, E, nullptr, kbuf, nullptr, S, nullptr, 0);
    gemm_bf16_wmma<<<gE, 32, 0, stream>>>(h1, wvt, M, E, E, nullptr, nullptr, vtb, S, nullptr, 0);

    // 4) causal flash attention
    dim3 ga(S / 16, H, B);                   // (128, 8, 4)
    attn_flash_kernel<<<ga, 32, 0, stream>>>(qbuf, kbuf, vtb, attn, S, E);

    // 5) output projections; residual uses the ORIGINAL block input x
    gemm_bf16_wmma<<<gE, 32, 0, stream>>>(attn, wo1t, M, E, E, nullptr, p1, nullptr, S, nullptr, 0);
    gemm_bf16_wmma<<<gE, 32, 0, stream>>>(p1, wo2t, M, E, E, res1, nullptr, nullptr, S, x, 0);

    // 6) LN2 + MLP
    ln_bf16_kernel<<<M, 256, 0, stream>>>(res1, ln2g, h2, E);
    dim3 gE4(E4 / 16, M / 64);               // (128, 128)
    gemm_bf16_wmma<<<gE4, 32, 0, stream>>>(h2, w1t, M, E4, E, nullptr, mlp1, nullptr, S, nullptr, 1);
    gemm_bf16_wmma<<<gE, 32, 0, stream>>>(mlp1, w2t, M, E, E4, out, nullptr, nullptr, S, res1, 0);
}
